// VectorQuantizer_137438954121
// MI455X (gfx1250) — matrix operand order repaired and checked by reading, not compiled, not run on hardware
//
#include <hip/hip_runtime.h>

// ---------------------------------------------------------------------------
// Vector-quantizer: N=32768 rows (C=256) vs K=8192 codes.
// score[n,k] = ||e_k||^2 - 2 * x_n . e_k   (||x||^2 constant per row)
// Matmul done with split-bf16 emulation: x=xh+xl, e=eh+el,
//   x.e ~= xh.eh + xh.el + xl.eh   (3x v_wmma_f32_16x16x32_bf16 per K=32)
// ---------------------------------------------------------------------------

typedef __attribute__((ext_vector_type(16))) __bf16 v16bf;
typedef __attribute__((ext_vector_type(8)))  float  v8f;

union BF16Tile {           // one WMMA 16-bit operand: 16 bf16 per lane = 32 B
    v16bf v;
    uint4 q[2];
};

#define VQ_N 32768
#define VQ_C 256
#define VQ_K 8192

__device__ __forceinline__ unsigned short f32_to_bf16_rne(float f) {
    unsigned u = __float_as_uint(f);
    unsigned r = u + 0x7FFFu + ((u >> 16) & 1u);
    return (unsigned short)(r >> 16);
}
__device__ __forceinline__ float bf16_bits_to_f32(unsigned short h) {
    return __uint_as_float(((unsigned)h) << 16);
}

// --------------------------------------------------------------------------
// Prep 1: decompose weights into bf16 hi/lo planes + compute ||e_k||^2.
// One block (256 threads) per code row.
// --------------------------------------------------------------------------
__global__ void vq_prep_w(const float* __restrict__ W,
                          unsigned short* __restrict__ Whi,
                          unsigned short* __restrict__ Wlo,
                          float* __restrict__ eNorm) {
    const int k = blockIdx.x;
    const int c = threadIdx.x;
    const size_t i = (size_t)k * VQ_C + c;
    float w = W[i];
    unsigned short h = f32_to_bf16_rne(w);
    float lo = w - bf16_bits_to_f32(h);
    Whi[i] = h;
    Wlo[i] = f32_to_bf16_rne(lo);

    // reduce sum(w*w) over the 256-thread block
    float s = w * w;
    #pragma unroll
    for (int off = 16; off >= 1; off >>= 1)
        s += __shfl_xor(s, off, 32);
    __shared__ float part[8];
    if ((threadIdx.x & 31) == 0) part[threadIdx.x >> 5] = s;
    __syncthreads();
    if (threadIdx.x == 0) {
        float t = 0.f;
        #pragma unroll
        for (int j = 0; j < 8; ++j) t += part[j];
        eNorm[k] = t;
    }
}

// --------------------------------------------------------------------------
// Prep 2: decompose inputs into bf16 hi/lo planes. 4 floats per thread.
// --------------------------------------------------------------------------
__global__ void vq_prep_x(const float* __restrict__ X,
                          unsigned short* __restrict__ Xhi,
                          unsigned short* __restrict__ Xlo) {
    const size_t i = ((size_t)blockIdx.x * blockDim.x + threadIdx.x) * 4;
    float4 v = *(const float4*)(X + i);
    unsigned short h0 = f32_to_bf16_rne(v.x), h1 = f32_to_bf16_rne(v.y);
    unsigned short h2 = f32_to_bf16_rne(v.z), h3 = f32_to_bf16_rne(v.w);
    unsigned short l0 = f32_to_bf16_rne(v.x - bf16_bits_to_f32(h0));
    unsigned short l1 = f32_to_bf16_rne(v.y - bf16_bits_to_f32(h1));
    unsigned short l2 = f32_to_bf16_rne(v.z - bf16_bits_to_f32(h2));
    unsigned short l3 = f32_to_bf16_rne(v.w - bf16_bits_to_f32(h3));
    uint2 hv, lv;
    hv.x = (unsigned)h0 | ((unsigned)h1 << 16);
    hv.y = (unsigned)h2 | ((unsigned)h3 << 16);
    lv.x = (unsigned)l0 | ((unsigned)l1 << 16);
    lv.y = (unsigned)l2 | ((unsigned)l3 << 16);
    *(uint2*)(Xhi + i) = hv;
    *(uint2*)(Xlo + i) = lv;
}

// --------------------------------------------------------------------------
// Main: fused GEMM + argmin.
// 256 blocks x 256 threads (8 waves). Wave w owns rows blockIdx*128 + w*16.
// For each chunk of 16 codes: 8 K-tiles x 3 bf16 WMMAs -> 16x16 f32 scores.
// A layout (16-bit 16x32): lane L -> M=L&15, half=L>>4; elem p -> K = p + 8h + (p>=8?8:0)
// B layout (16-bit 32x16): lane L -> N=L&15, half=L>>4; elem p -> K = p + 8h + (p>=8?8:0)
// C/D layout (f32 16x16):  lane L -> N=L&15; VGPR r -> M = r + 8*(L>>4)
// --------------------------------------------------------------------------
__global__ __launch_bounds__(256, 1)
void vq_argmin(const unsigned short* __restrict__ Xhi,
               const unsigned short* __restrict__ Xlo,
               const unsigned short* __restrict__ Whi,
               const unsigned short* __restrict__ Wlo,
               const float* __restrict__ eNorm,
               int* __restrict__ idxOut,
               float* __restrict__ idxOutF) {
    const int lane = threadIdx.x & 31;
    const int wave = threadIdx.x >> 5;
    const int half = lane >> 4;
    const int l15  = lane & 15;
    const int rowBase = blockIdx.x * 128 + wave * 16;

    // Preload this wave's A operands (hi and lo) for all 8 K-tiles.
    BF16Tile Ahi[8], Alo[8];
    const uint4* xh = (const uint4*)(Xhi + (size_t)(rowBase + l15) * VQ_C);
    const uint4* xl = (const uint4*)(Xlo + (size_t)(rowBase + l15) * VQ_C);
    #pragma unroll
    for (int t = 0; t < 8; ++t) {
        Ahi[t].q[0] = xh[t * 4 + half];
        Ahi[t].q[1] = xh[t * 4 + half + 2];
        Alo[t].q[0] = xl[t * 4 + half];
        Alo[t].q[1] = xl[t * 4 + half + 2];
    }

    float minVal[8];
    int   minIdx[8];
    #pragma unroll
    for (int r = 0; r < 8; ++r) { minVal[r] = 3.4e38f; minIdx[r] = 0; }

    for (int cb = 0; cb < VQ_K; cb += 16) {
        const int code = cb + l15;
        const uint4* wh = (const uint4*)(Whi + (size_t)code * VQ_C);
        const uint4* wl = (const uint4*)(Wlo + (size_t)code * VQ_C);

        v8f acc = {0.f, 0.f, 0.f, 0.f, 0.f, 0.f, 0.f, 0.f};
        #pragma unroll
        for (int t = 0; t < 8; ++t) {
            BF16Tile bh, bl;
            bh.q[0] = wh[t * 4 + half];
            bh.q[1] = wh[t * 4 + half + 2];
            bl.q[0] = wl[t * 4 + half];
            bl.q[1] = wl[t * 4 + half + 2];
            acc = __builtin_amdgcn_wmma_f32_16x16x32_bf16(
                false, Ahi[t].v, false, bh.v, (short)0, acc, false, false);
            acc = __builtin_amdgcn_wmma_f32_16x16x32_bf16(
                false, Ahi[t].v, false, bl.v, (short)0, acc, false, false);
            acc = __builtin_amdgcn_wmma_f32_16x16x32_bf16(
                false, Alo[t].v, false, bh.v, (short)0, acc, false, false);
        }

        const float e2 = eNorm[code];
        #pragma unroll
        for (int r = 0; r < 8; ++r) {
            float s = fmaf(-2.0f, acc[r], e2);
            if (s < minVal[r]) { minVal[r] = s; minIdx[r] = code; }
        }
    }

    // Cross-lane argmin within each 16-lane half (xor masks < 16 stay in-half).
    #pragma unroll
    for (int r = 0; r < 8; ++r) {
        #pragma unroll
        for (int off = 1; off < 16; off <<= 1) {
            float ov = __shfl_xor(minVal[r], off, 32);
            int   oi = __shfl_xor(minIdx[r], off, 32);
            if (ov < minVal[r] || (ov == minVal[r] && oi < minIdx[r])) {
                minVal[r] = ov; minIdx[r] = oi;
            }
        }
        if (l15 == 0) {
            int row = rowBase + half * 8 + r;
            idxOut[row]  = minIdx[r];
            idxOutF[row] = (float)minIdx[r];
        }
    }
}

// --------------------------------------------------------------------------
// Gather + straight-through arithmetic: out = x + (w[idx] - x), f32.
// 4 rows per 256-thread block, float4 per thread.
// --------------------------------------------------------------------------
__global__ void vq_gather(const float* __restrict__ W,
                          const float* __restrict__ X,
                          const int* __restrict__ idx,
                          float* __restrict__ out) {
    const int tid = threadIdx.x;
    const int rowInBlk = tid >> 6;   // 64 threads per row
    const int c4 = tid & 63;
    const int row = blockIdx.x * 4 + rowInBlk;
    const int k = idx[row];
    float4 q = ((const float4*)(W + (size_t)k * VQ_C))[c4];
    float4 x = ((const float4*)(X + (size_t)row * VQ_C))[c4];
    float4 o;
    o.x = x.x + (q.x - x.x);
    o.y = x.y + (q.y - x.y);
    o.z = x.z + (q.z - x.z);
    o.w = x.w + (q.w - x.w);
    ((float4*)(out + (size_t)row * VQ_C))[c4] = o;
}

// --------------------------------------------------------------------------
extern "C" void kernel_launch(void* const* d_in, const int* in_sizes, int n_in,
                              void* d_out, int out_size, void* d_ws, size_t ws_size,
                              hipStream_t stream) {
    const float* X = (const float*)d_in[0];   // (8,4096,256) f32
    const float* W = (const float*)d_in[1];   // (8192,256)   f32
    float* out     = (float*)d_out;           // quantized then indices

    char* ws = (char*)d_ws;
    const size_t XHI = 0;
    const size_t XLO = XHI + (size_t)VQ_N * VQ_C * 2;   // +16 MB
    const size_t WHI = XLO + (size_t)VQ_N * VQ_C * 2;   // +16 MB
    const size_t WLO = WHI + (size_t)VQ_K * VQ_C * 2;   // +4 MB
    const size_t EN  = WLO + (size_t)VQ_K * VQ_C * 2;   // +4 MB
    const size_t IDX = EN  + (size_t)VQ_K * 4;          // +32 KB

    unsigned short* Xhi = (unsigned short*)(ws + XHI);
    unsigned short* Xlo = (unsigned short*)(ws + XLO);
    unsigned short* Whi = (unsigned short*)(ws + WHI);
    unsigned short* Wlo = (unsigned short*)(ws + WLO);
    float* eNorm = (float*)(ws + EN);
    int*   idxWs = (int*)(ws + IDX);

    float* quantized = out;
    float* idxF = out + (size_t)VQ_N * VQ_C;

    vq_prep_w<<<VQ_K, 256, 0, stream>>>(W, Whi, Wlo, eNorm);
    vq_prep_x<<<(VQ_N * VQ_C) / (256 * 4), 256, 0, stream>>>(X, Xhi, Xlo);
    vq_argmin<<<VQ_N / 128, 256, 0, stream>>>(Xhi, Xlo, Whi, Wlo, eNorm, idxWs, idxF);
    vq_gather<<<VQ_N / 4, 256, 0, stream>>>(W, X, idxWs, quantized);
}
